// COREAttentiveCollaboration_83597243450090
// MI455X (gfx1250) — compile-verified, matrix operand order checked by reading, not run on hardware
//
#include <hip/hip_runtime.h>
#include <math.h>

#define N_AG 6
#define C_CH 64
#define HH 128
#define WW 128
#define HW (HH*WW)

typedef __attribute__((ext_vector_type(16))) _Float16 v16h;
typedef __attribute__((ext_vector_type(8)))  _Float16 v8h;
typedef __attribute__((ext_vector_type(8)))  float    v8f;

// ---- workspace layout (bytes, all offsets 64B aligned) ----
#define OFF_CONF   0u          // 6*16384 f32            = 393216
#define OFF_MM     393216u     // 12 uint (min/max bits) -> pad 64
#define OFF_WSTACK 393280u     // 192*64 f16             = 24576
#define OFF_PROJW  417856u     // 64*64 f16              = 8192
#define OFF_RAWT   426048u     // [i][p][c] f16          = 12582912
#define OFF_P      13008960u   // [i][c][p] f16          = 12582912
#define OFF_Q      25591872u   // [i][c][p] f16          = 12582912
#define OFF_VPROJ  38174784u   // [k][p][c] f16          = 12582912
#define OFF_SSUM   50757696u   // [i][p][c] f16          = 12582912
// total ~63.3 MB

__device__ __forceinline__ float gelu_exact(float x) {
    return 0.5f * x * (1.0f + erff(x * 0.7071067811865475f));
}

// ---------------- K0a: raw f32 [i][c][p] -> f16 pixel-major [i][p][c] ----------------
__global__ void k_convert_raw(const float* __restrict__ raw, _Float16* __restrict__ rawt) {
    int i = blockIdx.y;
    int p = blockIdx.x * 256 + threadIdx.x;
    const float* src = raw + (size_t)i * C_CH * HW + p;
    _Float16* dst = rawt + ((size_t)i * HW + p) * C_CH;
#pragma unroll
    for (int v = 0; v < 8; ++v) {
        v8h pack;
#pragma unroll
        for (int t = 0; t < 8; ++t) pack[t] = (_Float16)src[(size_t)(v * 8 + t) * HW];
        *(v8h*)(dst + v * 8) = pack;
    }
}

// ---------------- K0b: pack [W_a; W_b; W_v] (192x64) and proj (64x64) to f16 ----------------
__global__ void k_pack_weights(const float* __restrict__ a1w, const float* __restrict__ vw,
                               const float* __restrict__ prw,
                               _Float16* __restrict__ wstack, _Float16* __restrict__ projw) {
    for (int idx = threadIdx.x; idx < 192 * 64; idx += blockDim.x) {
        int r = idx >> 6, c = idx & 63;
        float v;
        if (r < 64)       v = a1w[r * 128 + c];             // W_a = cm_a1_w[:, :C]  (ego i)
        else if (r < 128) v = a1w[(r - 64) * 128 + 64 + c]; // W_b = cm_a1_w[:, C:]  (sender k)
        else              v = vw[(r - 128) * 64 + c];
        wstack[idx] = (_Float16)v;
    }
    for (int idx = threadIdx.x; idx < 64 * 64; idx += blockDim.x)
        projw[idx] = (_Float16)prw[idx];
}

// ---------------- K_init: min/max slots ----------------
__global__ void k_init_mm(unsigned* mm) {
    int t = threadIdx.x;
    if (t < N_AG) { mm[2 * t] = 0x7F800000u; mm[2 * t + 1] = 0u; }
}

// ---------------- K1a: confidence pre-map + per-agent min/max ----------------
__global__ void k_conf(const float* __restrict__ raw,
                       const float* __restrict__ w1, const float* __restrict__ b1,
                       const float* __restrict__ gam, const float* __restrict__ bet,
                       const float* __restrict__ mean, const float* __restrict__ var,
                       const float* __restrict__ w2, const float* __restrict__ b2,
                       float* __restrict__ conf, unsigned* __restrict__ mm) {
    __shared__ float w1s[4 * 64];
    __shared__ float rmin[256], rmax[256];
    int tid = threadIdx.x;
    w1s[tid] = w1[tid];
    __syncthreads();
    int i = blockIdx.y;
    int p = blockIdx.x * 256 + tid;
    const float* src = raw + (size_t)i * C_CH * HW + p;
    float h0 = 0.f, h1 = 0.f, h2 = 0.f, h3 = 0.f;
    for (int c = 0; c < C_CH; ++c) {
        float x = src[(size_t)c * HW];
        h0 += w1s[c] * x; h1 += w1s[64 + c] * x;
        h2 += w1s[128 + c] * x; h3 += w1s[192 + c] * x;
    }
    float h[4] = {h0 + b1[0], h1 + b1[1], h2 + b1[2], h3 + b1[3]};
    float acc = b2[0];
#pragma unroll
    for (int j = 0; j < 4; ++j) {
        float s = gam[j] / sqrtf(var[j] + 1e-5f);
        float t = (h[j] - mean[j]) * s + bet[j];
        t = fmaxf(t, 0.0f);
        acc += w2[j] * t;
    }
    acc = fmaxf(acc, 0.0f);          // >= 0  -> uint bit order == float order
    conf[i * HW + p] = acc;
    rmin[tid] = acc; rmax[tid] = acc;
    __syncthreads();
    for (int s = 128; s > 0; s >>= 1) {
        if (tid < s) {
            rmin[tid] = fminf(rmin[tid], rmin[tid + s]);
            rmax[tid] = fmaxf(rmax[tid], rmax[tid + s]);
        }
        __syncthreads();
    }
    if (tid == 0) {
        atomicMin(&mm[2 * i],     __float_as_uint(rmin[0]));
        atomicMax(&mm[2 * i + 1], __float_as_uint(rmax[0]));
    }
}

// ---------------- K1b: min-max normalize ----------------
__global__ void k_norm(float* __restrict__ conf, const unsigned* __restrict__ mm) {
    int i = blockIdx.y;
    int p = blockIdx.x * 256 + threadIdx.x;
    float mn = __uint_as_float(mm[2 * i]);
    float mx = __uint_as_float(mm[2 * i + 1]);
    float inv = 1.0f / (mx - mn);
    conf[i * HW + p] = (conf[i * HW + p] - mn) * inv;
}

// ---- shared WMMA micro-GEMM: one wave, D[16x16] = A[16x64] x B[64x16], f16 in / f32 acc ----
__device__ __forceinline__ v8f wmma_16x16x64(const _Float16* __restrict__ wrow,
                                             const _Float16* __restrict__ brow, int g) {
    v8f acc = {};
#pragma unroll
    for (int kk = 0; kk < 64; kk += 32) {
        v16h A, B;
        v8h alo = *(const v8h*)(wrow + kk + 8 * g);
        v8h ahi = *(const v8h*)(wrow + kk + 16 + 8 * g);
        v8h blo = *(const v8h*)(brow + kk + 8 * g);
        v8h bhi = *(const v8h*)(brow + kk + 16 + 8 * g);
#pragma unroll
        for (int t = 0; t < 8; ++t) {
            A[t] = alo[t]; A[8 + t] = ahi[t];
            B[t] = blo[t]; B[8 + t] = bhi[t];
        }
        acc = __builtin_amdgcn_wmma_f32_16x16x32_f16(false, A, false, B, (short)0, acc,
                                                     false, false);
    }
    return acc;
}

// ---------------- K2: stacked WMMA GEMM  D[192,16K] = Wstack[192,64] x raw[i][64,16K] ----------------
// one wave per 16x16 tile; destination (P / Q / vproj) is uniform per rb -> scalar branch
__global__ void k_gemm_stack(const _Float16* __restrict__ rawt,
                             const _Float16* __restrict__ wstack,
                             const float* __restrict__ vb,
                             _Float16* __restrict__ P,
                             _Float16* __restrict__ Q,
                             _Float16* __restrict__ vproj) {
    int lane = threadIdx.x;
    int g = lane >> 4, n = lane & 15;
    int tile = blockIdx.x, rb = blockIdx.y, i = blockIdx.z;
    int p = tile * 16 + n;
    const _Float16* wrow = wstack + (size_t)(rb * 16 + n) * 64;   // A row M=n
    const _Float16* brow = rawt + ((size_t)i * HW + p) * C_CH;    // B col N=n
    v8f acc = wmma_16x16x64(wrow, brow, g);

    if (rb < 4) {                       // rows [0,64)   -> P[i][c][p]
        _Float16* dst = P + (size_t)i * C_CH * HW + (size_t)(rb * 16 + 8 * g) * HW + p;
#pragma unroll
        for (int j = 0; j < 8; ++j) dst[(size_t)j * HW] = (_Float16)acc[j];
    } else if (rb < 8) {                // rows [64,128) -> Q[i][c][p]
        _Float16* dst = Q + (size_t)i * C_CH * HW + (size_t)((rb - 4) * 16 + 8 * g) * HW + p;
#pragma unroll
        for (int j = 0; j < 8; ++j) dst[(size_t)j * HW] = (_Float16)acc[j];
    } else {                            // rows [128,192) -> vproj[i][p][o] (+bias)
        int ob = (rb - 8) * 16 + 8 * g;
        _Float16* dst = vproj + ((size_t)i * HW + p) * C_CH + ob;
#pragma unroll
        for (int j = 0; j < 8; ++j) dst[j] = (_Float16)(acc[j] + vb[ob + j]);
    }
}

// ---------------- K3: fused gelu + 11x11 depthwise + att/vproj scale + sum over senders ----------------
__global__ __launch_bounds__(256)
void k_fuse(const _Float16* __restrict__ P, const _Float16* __restrict__ Q,
            const _Float16* __restrict__ vproj, const float* __restrict__ conf,
            const float* __restrict__ a1b, const float* __restrict__ dww,
            const float* __restrict__ dwb, _Float16* __restrict__ Ssum) {
    __shared__ float s_apre[8][26][26];     // 8-channel group, 16 tile + 2*5 halo
    __shared__ float s_dww[64 * 121];
    __shared__ float s_a1b[64];
    __shared__ float s_dwb[64];
    int tid = threadIdx.x;
    int i = blockIdx.z;
    int tx = blockIdx.x, ty = blockIdx.y;
    for (int idx = tid; idx < 64 * 121; idx += 256) s_dww[idx] = dww[idx];
    if (tid < 64) { s_a1b[tid] = a1b[tid]; s_dwb[tid] = dwb[tid]; }
    int px = tid & 15, py = tid >> 4;
    int gx = tx * 16 + px, gy = ty * 16 + py;
    int p = gy * WW + gx;
    float S[64];
#pragma unroll
    for (int c = 0; c < 64; ++c) S[c] = 0.f;
    float wi = 1.0f - conf[i * HW + p];
    __syncthreads();
    for (int k = 0; k < N_AG; ++k) {
        if (k + 1 < N_AG) {             // prefetch next sender's activation region
            __builtin_prefetch(Q + (size_t)(k + 1) * C_CH * HW + p, 0, 1);
            __builtin_prefetch(vproj + ((size_t)(k + 1) * HW + p) * C_CH, 0, 1);
        }
        float att = wi * conf[k * HW + p];
        const _Float16* vrow = vproj + ((size_t)k * HW + p) * C_CH;
        const _Float16* Pb = P + (size_t)i * C_CH * HW;
        const _Float16* Qb = Q + (size_t)k * C_CH * HW;
#pragma unroll
        for (int cg = 0; cg < 8; ++cg) {
            __syncthreads();                       // protect previous group's reads
            for (int idx = tid; idx < 8 * 26 * 26; idx += 256) {
                int c = idx / 676; int rem = idx - c * 676;
                int yy = rem / 26; int xx = rem - yy * 26;
                int sy = ty * 16 + yy - 5, sx = tx * 16 + xx - 5;
                float v = 0.f;
                if (sy >= 0 && sy < HH && sx >= 0 && sx < WW) {
                    int cc = cg * 8 + c;
                    size_t o = (size_t)cc * HW + (size_t)sy * WW + sx;
                    v = gelu_exact((float)Pb[o] + (float)Qb[o] + s_a1b[cc]);
                }
                s_apre[c][yy][xx] = v;
            }
            __syncthreads();
            v8h vp = *(const v8h*)(vrow + cg * 8);
#pragma unroll
            for (int c = 0; c < 8; ++c) {
                int cc = cg * 8 + c;
                float acc = s_dwb[cc];
                const float* wv = &s_dww[cc * 121];
#pragma unroll
                for (int dy = 0; dy < 11; ++dy)
#pragma unroll
                    for (int dx = 0; dx < 11; ++dx)
                        acc += wv[dy * 11 + dx] * s_apre[c][py + dy][px + dx];
                S[cc] += acc * (float)vp[c] * att;
            }
        }
    }
    _Float16* srow = Ssum + ((size_t)i * HW + p) * C_CH;
#pragma unroll
    for (int v = 0; v < 8; ++v) {
        v8h pack;
#pragma unroll
        for (int t = 0; t < 8; ++t) pack[t] = (_Float16)S[v * 8 + t];
        *(v8h*)(srow + v * 8) = pack;
    }
}

// ---------------- K4: WMMA proj GEMM + residual + 6*bias ----------------
__global__ void k_proj(const _Float16* __restrict__ Ssum, const _Float16* __restrict__ projw,
                       const float* __restrict__ raw, const float* __restrict__ pb,
                       float* __restrict__ out) {
    int lane = threadIdx.x;
    int g = lane >> 4, n = lane & 15;
    int tile = blockIdx.x, rb = blockIdx.y, i = blockIdx.z;
    int p = tile * 16 + n;
    const _Float16* wrow = projw + (size_t)(rb * 16 + n) * 64;
    const _Float16* brow = Ssum + ((size_t)i * HW + p) * C_CH;
    v8f acc = wmma_16x16x64(wrow, brow, g);

    int r0 = rb * 16 + 8 * g;
    const float* rsrc = raw + ((size_t)i * C_CH + r0) * HW + p;
    float* dst = out + ((size_t)i * C_CH + r0) * HW + p;
#pragma unroll
    for (int j = 0; j < 8; ++j)
        dst[(size_t)j * HW] = rsrc[(size_t)j * HW] + acc[j] + 6.0f * pb[r0 + j];
}

extern "C" void kernel_launch(void* const* d_in, const int* in_sizes, int n_in,
                              void* d_out, int out_size, void* d_ws, size_t ws_size,
                              hipStream_t stream) {
    const float* raw   = (const float*)d_in[0];
    const float* pw_w1 = (const float*)d_in[1];
    const float* pw_b1 = (const float*)d_in[2];
    const float* bn_g  = (const float*)d_in[3];
    const float* bn_b  = (const float*)d_in[4];
    const float* bn_m  = (const float*)d_in[5];
    const float* bn_v  = (const float*)d_in[6];
    const float* pw_w2 = (const float*)d_in[7];
    const float* pw_b2 = (const float*)d_in[8];
    const float* a1_w  = (const float*)d_in[9];
    const float* a1_b  = (const float*)d_in[10];
    const float* dw_w  = (const float*)d_in[11];
    const float* dw_b  = (const float*)d_in[12];
    const float* v_w   = (const float*)d_in[13];
    const float* v_b   = (const float*)d_in[14];
    const float* pr_w  = (const float*)d_in[15];
    const float* pr_b  = (const float*)d_in[16];
    float* out = (float*)d_out;

    char* ws = (char*)d_ws;
    float*     conf   = (float*)(ws + OFF_CONF);
    unsigned*  mm     = (unsigned*)(ws + OFF_MM);
    _Float16*  wstack = (_Float16*)(ws + OFF_WSTACK);
    _Float16*  projw  = (_Float16*)(ws + OFF_PROJW);
    _Float16*  rawt   = (_Float16*)(ws + OFF_RAWT);
    _Float16*  Pbuf   = (_Float16*)(ws + OFF_P);
    _Float16*  Qbuf   = (_Float16*)(ws + OFF_Q);
    _Float16*  vproj  = (_Float16*)(ws + OFF_VPROJ);
    _Float16*  ssum   = (_Float16*)(ws + OFF_SSUM);

    k_convert_raw<<<dim3(64, 6), 256, 0, stream>>>(raw, rawt);
    k_pack_weights<<<1, 256, 0, stream>>>(a1_w, v_w, pr_w, wstack, projw);
    k_init_mm<<<1, 32, 0, stream>>>(mm);
    k_conf<<<dim3(64, 6), 256, 0, stream>>>(raw, pw_w1, pw_b1, bn_g, bn_b, bn_m, bn_v,
                                            pw_w2, pw_b2, conf, mm);
    k_norm<<<dim3(64, 6), 256, 0, stream>>>(conf, mm);
    k_gemm_stack<<<dim3(1024, 12, 6), 32, 0, stream>>>(rawt, wstack, v_b, Pbuf, Qbuf, vproj);
    k_fuse<<<dim3(8, 8, 6), 256, 0, stream>>>(Pbuf, Qbuf, vproj, conf, a1_b, dw_w, dw_b, ssum);
    k_proj<<<dim3(1024, 4, 6), 32, 0, stream>>>(ssum, projw, raw, pr_b, out);
}
